// AttNHPEBM_58703613002164
// MI455X (gfx1250) — compile-verified, matrix operand order checked by reading, not run on hardware
//
#include <hip/hip_runtime.h>
#include <hip/hip_bf16.h>

typedef __attribute__((ext_vector_type(16))) _Float16 v16h;
typedef __attribute__((ext_vector_type(8)))  float    v8f;

#define NSEQ 512     // B*S
#define LSEQ 128     // L
#define TWO_L 256
#define DM 256
#define DT 32
#define DIN 288      // DM + DT
#define EMB3 384     // 3*EMB_DIM
#define DFEAT 512    // DM * N_HEADS

__device__ __forceinline__ v8f wmma16(v16h a, v16h b, v8f c) {
    return __builtin_amdgcn_wmma_f32_16x16x32_f16(false, a, false, b, (short)0, c, false, false);
}

// GEMM chunk: preload KC B-fragments, then WMMA chain.
template <int KC>
__device__ __forceinline__ v8f gemm_chunk(const v16h* a, const _Float16* wrow, v8f acc) {
    v16h b[KC];
#pragma unroll
    for (int i = 0; i < KC; ++i) b[i] = *(const v16h*)(wrow + i * 32);
#pragma unroll
    for (int i = 0; i < KC; ++i) acc = wmma16(a[i], b[i], acc);
    return acc;
}

// bulk async global->LDS copy: nbytes must be multiple of 16*nthreads
__device__ __forceinline__ void async_fill(int tid, int nthreads, const void* gsrc,
                                           void* ldst, int nbytes) {
    unsigned int ldsbase = (unsigned int)(uintptr_t)ldst;
    const char* g = (const char*)gsrc;
    for (int i = tid; i < nbytes / 16; i += nthreads) {
        asm volatile("global_load_async_to_lds_b128 %0, %1, off"
                     :: "v"(ldsbase + (unsigned int)i * 16u), "v"(g + (size_t)i * 16)
                     : "memory");
    }
}

// ---------------- weight convert (f32 KxN -> f16 NxK transposed) ----------------
__global__ void attnhp_cvt_tr(const float* __restrict__ src, _Float16* __restrict__ dst,
                              int K, int N) {
    int i = blockIdx.x * blockDim.x + threadIdx.x;
    if (i < K * N) {
        int kk = i / N, nn = i % N;
        dst[(size_t)nn * K + kk] = (_Float16)src[i];
    }
}

// ---------------- embedding gather + input projection ----------------
__global__ void attnhp_embed(const int* __restrict__ subj, const int* __restrict__ obj,
                             const int* __restrict__ rel, const float* __restrict__ timev,
                             const float* __restrict__ ent, const float* __restrict__ relemb,
                             const _Float16* __restrict__ winw_t, const float* __restrict__ inb,
                             _Float16* __restrict__ enc, _Float16* __restrict__ tembuf) {
    __shared__ _Float16 As[16 * EMB3];
    const int tok0 = blockIdx.x * 16;      // flat token = n*L + l
    const int tid = threadIdx.x;           // 128 threads
    for (int i = tid; i < 16 * EMB3; i += 128) {
        int r = i / EMB3, c = i % EMB3;
        int tok = tok0 + r;
        float v;
        if (c < 128)       v = ent[(size_t)subj[tok] * 128 + c];
        else if (c < 256)  v = ent[(size_t)obj[tok] * 128 + (c - 128)];
        else               v = relemb[(size_t)rel[tok] * 128 + (c - 256)];
        As[i] = (_Float16)v;
    }
    if (tid < 16) {
        int tok = tok0 + tid;
        float t = timev[tok];
        float np = t > 0.f ? 1.f : 0.f;
        _Float16* erow = enc + (size_t)tok * DIN + DM;
        _Float16* trow = tembuf + (size_t)tok * DT;
        for (int i = 0; i < 16; ++i) {
            float div = __expf((float)(2 * i) * (-9.210340371976184f / 32.f));
            float a = t * div;
            _Float16 s = (_Float16)(__sinf(a) * np);
            _Float16 c = (_Float16)(__cosf(a) * np);
            erow[2 * i] = s; erow[2 * i + 1] = c;
            trow[2 * i] = s; trow[2 * i + 1] = c;
        }
    }
    __syncthreads();
    const int wave = tid >> 5, lane = tid & 31;
    const int lr = lane & 15, lh = lane >> 4;
    v16h a[12];
#pragma unroll
    for (int kc = 0; kc < 12; ++kc)
        a[kc] = *(const v16h*)(As + lr * EMB3 + kc * 32 + (lh << 4));
    for (int nt = wave; nt < 16; nt += 4) {
        int nbase = nt * 16;
        const _Float16* wrow = winw_t + (size_t)(nbase + lr) * EMB3 + (lh << 4);
        v8f acc = {};
        acc = gemm_chunk<6>(a, wrow, acc);
        acc = gemm_chunk<6>(a + 6, wrow + 6 * 32, acc);
        int colw = nbase + lr;
        float bias = inb[colw];
        for (int j = 0; j < 8; ++j) {
            int row = j + (lh << 3);
            enc[(size_t)(tok0 + row) * DIN + colw] = (_Float16)(acc[j] + bias);
        }
    }
}

// ---------------- QKV projection for one (head, layer) ----------------
// 256 threads; weights staged in LDS via async copy, double-buffered across q/k/v.
#define QKV_WBYTES (DM * DIN * 2)                       // 147456 B per matrix
#define QKV_SMEM   (16 * DIN * 2 + 2 * QKV_WBYTES)      // A tile + 2 weight buffers
#define QKV_FILL_OPS_PER_WAVE (QKV_WBYTES / 16 / 256)   // 36 async ops per wave

__global__ void attnhp_qkv(const _Float16* __restrict__ enc, const float* __restrict__ cur,
                           const _Float16* __restrict__ tembuf,
                           const _Float16* __restrict__ wqt, const _Float16* __restrict__ wkt,
                           const _Float16* __restrict__ wvt,
                           const float* __restrict__ qb, const float* __restrict__ kb,
                           const float* __restrict__ vb,
                           _Float16* __restrict__ qo, _Float16* __restrict__ ko,
                           _Float16* __restrict__ vo) {
    extern __shared__ char qsmem[];
    _Float16* As = (_Float16*)qsmem;                                   // 16 x 288
    _Float16* W0 = (_Float16*)(qsmem + 16 * DIN * 2);                  // 256 x 288
    _Float16* W1 = (_Float16*)(qsmem + 16 * DIN * 2 + QKV_WBYTES);     // 256 x 288
    const int n = blockIdx.x >> 4;
    const int row0 = (blockIdx.x & 15) * 16;   // row within 2L=256
    const int tid = threadIdx.x;               // 256

    // kick off async fill of first weight matrix while we build A
    async_fill(tid, 256, wqt, W0, QKV_WBYTES);

    for (int i = tid; i < 16 * DIN; i += 256) {
        int r = i / DIN, c = i % DIN;
        int row = row0 + r;
        _Float16 v;
        if (row < LSEQ) {
            v = enc[((size_t)n * LSEQ + row) * DIN + c];
        } else {
            int r2 = row - LSEQ;
            if (c < DM) v = (_Float16)cur[((size_t)n * LSEQ + r2) * DM + c];
            else        v = tembuf[((size_t)n * LSEQ + r2) * DT + (c - DM)];
        }
        As[i] = v;
    }
    __syncthreads();

    const int wave = tid >> 5, lane = tid & 31;
    const int lr = lane & 15, lh = lane >> 4;
    v16h a[9];
#pragma unroll
    for (int kc = 0; kc < 9; ++kc)
        a[kc] = *(const v16h*)(As + lr * DIN + kc * 32 + (lh << 4));

    const _Float16* wts[3] = { wqt, wkt, wvt };
    const float*    bss[3] = { qb, kb, vb };
    _Float16*       ous[3] = { qo, ko, vo };
    _Float16*       wbuf[2] = { W0, W1 };

    for (int m = 0; m < 3; ++m) {
        if (m < 2) {
            // prefetch next weight matrix into the other buffer (previous users
            // of that buffer finished at the trailing __syncthreads of m-1)
            async_fill(tid, 256, wts[m + 1], wbuf[(m + 1) & 1], QKV_WBYTES);
            // wait until only the just-issued fill remains outstanding
            asm volatile("s_wait_asynccnt %0" :: "i"(QKV_FILL_OPS_PER_WAVE) : "memory");
        } else {
            asm volatile("s_wait_asynccnt 0x0" ::: "memory");
        }
        __syncthreads();

        const _Float16* W = wbuf[m & 1];
        const float* bp = bss[m];
        _Float16* op = ous[m];
        for (int nt = wave; nt < 16; nt += 8) {
            int nbase = nt * 16;
            const _Float16* wrow = W + (size_t)(nbase + lr) * DIN + (lh << 4);
            v8f acc = {};
            acc = gemm_chunk<9>(a, wrow, acc);
            int colw = nbase + lr;
            float bias = bp[colw];
            for (int j = 0; j < 8; ++j) {
                int row = row0 + j + (lh << 3);
                op[((size_t)n * TWO_L + row) * DM + colw] = (_Float16)(acc[j] + bias);
            }
        }
        __syncthreads();   // all waves done reading W before it is refilled
    }
}

// ---------------- attention over one sequence (block = sequence) ----------------
__device__ __forceinline__ bool attnhp_masked(int qi, int kj, const float* tv) {
    int qr = qi < LSEQ ? qi : qi - LSEQ;
    if (kj < LSEQ) return (kj >= qr) || (tv[kj] <= 0.f);
    return (qi < LSEQ) ? true : ((kj - LSEQ) != qr);
}

#define ATT_SMEM (TWO_L * DM * 2 * 2 + LSEQ * 4 + 8 * 16 * 32 * 2)

__global__ void attnhp_attn(const _Float16* __restrict__ q, const _Float16* __restrict__ k,
                            const _Float16* __restrict__ v, const float* __restrict__ timev,
                            _Float16* __restrict__ enc, float* __restrict__ cur) {
    extern __shared__ char smem_raw[];
    _Float16* Ks = (_Float16*)smem_raw;            // [key][d]   256x256
    _Float16* Vt = Ks + TWO_L * DM;                // [d][key]   256x256
    float*    tv = (float*)(Vt + TWO_L * DM);      // [L]
    _Float16* Ps = (_Float16*)(tv + LSEQ);         // [8][16][32] per-wave P scratch

    const int n = blockIdx.x;
    const int tid = threadIdx.x;                   // 256 threads
    const _Float16* kbase = k + (size_t)n * TWO_L * DM;
    const _Float16* vbase = v + (size_t)n * TWO_L * DM;
    {
        // K tile: CDNA5 async global->LDS copy (VGPR-bypassing, ASYNCcnt-tracked)
        async_fill(tid, 256, kbase, Ks, TWO_L * DM * 2);
        // V transpose (element-wise through VGPRs)
        for (int i = tid; i < TWO_L * DM; i += 256) {
            int key = i >> 8, d = i & 255;
            Vt[d * TWO_L + key] = vbase[i];
        }
        if (tid < LSEQ) tv[tid] = timev[(size_t)n * LSEQ + tid];
        asm volatile("s_wait_asynccnt 0x0" ::: "memory");
    }
    __syncthreads();

    const int wave = tid >> 5, lane = tid & 31;
    const int lr = lane & 15, lh = lane >> 4;
    _Float16* scratch = Ps + wave * 16 * 32;
    const float inv_sqrt_dk = 0.0625f;

    for (int qt = wave; qt < 16; qt += 8) {
        const int q0 = qt * 16;
        v16h qa[8];
        for (int dc = 0; dc < 8; ++dc)
            qa[dc] = *(const v16h*)(q + ((size_t)n * TWO_L + q0 + lr) * DM + dc * 32 + (lh << 4));

        // pass 1: row max
        float rmax[8];
        for (int j = 0; j < 8; ++j) rmax[j] = -3.4e38f;
        for (int kt = 0; kt < 16; ++kt) {
            const _Float16* krow = Ks + (kt * 16 + lr) * DM + (lh << 4);
            v8f acc = {};
            acc = gemm_chunk<8>(qa, krow, acc);
            int col = kt * 16 + lr;
            for (int j = 0; j < 8; ++j) {
                int qi = q0 + j + (lh << 3);
                float s = attnhp_masked(qi, col, tv) ? -1.0e9f : acc[j] * inv_sqrt_dk;
                rmax[j] = fmaxf(rmax[j], s);
            }
        }
        for (int m = 1; m < 16; m <<= 1)
            for (int j = 0; j < 8; ++j) rmax[j] = fmaxf(rmax[j], __shfl_xor(rmax[j], m));

        // pass 2: exp, sum, P@V
        float rsum[8];
        for (int j = 0; j < 8; ++j) rsum[j] = 0.f;
        v8f o[16];
        v8f zero = {};
        for (int ot = 0; ot < 16; ++ot) o[ot] = zero;
        for (int kp = 0; kp < 8; ++kp) {
            for (int half = 0; half < 2; ++half) {
                int kt = kp * 2 + half;
                const _Float16* krow = Ks + (kt * 16 + lr) * DM + (lh << 4);
                v8f acc = {};
                acc = gemm_chunk<8>(qa, krow, acc);
                int col = kt * 16 + lr;
                for (int j = 0; j < 8; ++j) {
                    int qi = q0 + j + (lh << 3);
                    float s = attnhp_masked(qi, col, tv) ? -1.0e9f : acc[j] * inv_sqrt_dk;
                    float p = __expf(s - rmax[j]);
                    rsum[j] += p;
                    scratch[(j + (lh << 3)) * 32 + half * 16 + lr] = (_Float16)p;
                }
            }
            v16h pa = *(const v16h*)(scratch + lr * 32 + (lh << 4));
            for (int ot = 0; ot < 16; ++ot) {
                v16h bf = *(const v16h*)(Vt + (ot * 16 + lr) * TWO_L + kp * 32 + (lh << 4));
                o[ot] = wmma16(pa, bf, o[ot]);
            }
        }
        for (int m = 1; m < 16; m <<= 1)
            for (int j = 0; j < 8; ++j) rsum[j] += __shfl_xor(rsum[j], m);

        // write enc_out: rows < L -> enc_input, rows >= L -> cur += tanh
        for (int ot = 0; ot < 16; ++ot) {
            int col = ot * 16 + lr;
            for (int j = 0; j < 8; ++j) {
                int qi = q0 + j + (lh << 3);
                float val = o[ot][j] / rsum[j];
                if (qi < LSEQ) {
                    enc[((size_t)n * LSEQ + qi) * DIN + col] = (_Float16)val;
                } else {
                    int r2 = qi - LSEQ;
                    float np = tv[r2] > 0.f ? 1.f : 0.f;
                    size_t ci = ((size_t)n * LSEQ + r2) * DM + col;
                    cur[ci] = tanhf(val * np) + cur[ci];
                }
            }
        }
    }
}

// ---------------- copy last-token cur into feat ----------------
__global__ void attnhp_head_out(const float* __restrict__ cur, float* __restrict__ feat, int h) {
    int i = blockIdx.x * blockDim.x + threadIdx.x;
    if (i < NSEQ * DM) {
        int n = i >> 8, c = i & 255;
        feat[(size_t)n * DFEAT + h * DM + c] = cur[((size_t)n * LSEQ + (LSEQ - 1)) * DM + c];
    }
}

__global__ void attnhp_init_out(float* __restrict__ out, const float* __restrict__ b2, int n) {
    int i = blockIdx.x * blockDim.x + threadIdx.x;
    if (i < n) out[i] = b2[0];
}

// ---------------- MLP head: gelu(feat @ w1 + b1) @ w2 ----------------
__global__ void attnhp_mlp(const float* __restrict__ feat, const _Float16* __restrict__ w1t,
                           const float* __restrict__ b1, const float* __restrict__ w2,
                           float* __restrict__ out) {
    __shared__ _Float16 As[16 * DFEAT];
    const int r0 = blockIdx.x * 16;
    const int tid = threadIdx.x;   // 128
    for (int i = tid; i < 16 * DFEAT; i += 128)
        As[i] = (_Float16)feat[(size_t)(r0 + i / DFEAT) * DFEAT + (i & (DFEAT - 1))];
    __syncthreads();
    const int wave = tid >> 5, lane = tid & 31;
    const int lr = lane & 15, lh = lane >> 4;
    v16h a[16];
#pragma unroll
    for (int kc = 0; kc < 16; ++kc)
        a[kc] = *(const v16h*)(As + lr * DFEAT + kc * 32 + (lh << 4));
    float part[8];
    for (int j = 0; j < 8; ++j) part[j] = 0.f;
    for (int nt = wave; nt < 16; nt += 4) {
        int nbase = nt * 16;
        const _Float16* wrow = w1t + (size_t)(nbase + lr) * DFEAT + (lh << 4);
        v8f acc = {};
        acc = gemm_chunk<8>(a, wrow, acc);
        acc = gemm_chunk<8>(a + 8, wrow + 8 * 32, acc);
        int col = nbase + lr;
        float bias = b1[col], wv = w2[col];
        for (int j = 0; j < 8; ++j) {
            float x = acc[j] + bias;
            float g = 0.5f * x * (1.f + erff(x * 0.70710678118f));
            part[j] += g * wv;
        }
    }
    for (int m = 1; m < 16; m <<= 1)
        for (int j = 0; j < 8; ++j) part[j] += __shfl_xor(part[j], m);
    if (lr == 0)
        for (int j = 0; j < 8; ++j)
            atomicAdd(&out[r0 + j + (lh << 3)], part[j]);
}

extern "C" void kernel_launch(void* const* d_in, const int* in_sizes, int n_in,
                              void* d_out, int out_size, void* d_ws, size_t ws_size,
                              hipStream_t stream) {
    const int*   subj   = (const int*)d_in[0];
    const int*   obj    = (const int*)d_in[1];
    const int*   rel    = (const int*)d_in[2];
    const float* timev  = (const float*)d_in[3];
    const float* ent    = (const float*)d_in[4];
    const float* relemb = (const float*)d_in[5];
    const float* in_w   = (const float*)d_in[6];
    const float* in_b   = (const float*)d_in[7];
    const float* q_w    = (const float*)d_in[8];
    const float* q_b    = (const float*)d_in[9];
    const float* k_w    = (const float*)d_in[10];
    const float* k_b    = (const float*)d_in[11];
    const float* v_w    = (const float*)d_in[12];
    const float* v_b    = (const float*)d_in[13];
    const float* w1     = (const float*)d_in[14];
    const float* b1     = (const float*)d_in[15];
    const float* w2     = (const float*)d_in[16];
    const float* b2     = (const float*)d_in[17];
    float* out = (float*)d_out;

    char* ws = (char*)d_ws;
    size_t off = 0;
    auto carve = [&](size_t bytes) -> char* {
        char* p = ws + off;
        off += (bytes + 255) & ~(size_t)255;
        return p;
    };
    _Float16* wInWt = (_Float16*)carve((size_t)DM * EMB3 * 2);
    _Float16* wQt   = (_Float16*)carve((size_t)4 * DM * DIN * 2);
    _Float16* wKt   = (_Float16*)carve((size_t)4 * DM * DIN * 2);
    _Float16* wVt   = (_Float16*)carve((size_t)4 * DM * DIN * 2);
    _Float16* wW1t  = (_Float16*)carve((size_t)DM * DFEAT * 2);
    _Float16* temb  = (_Float16*)carve((size_t)NSEQ * LSEQ * DT * 2);
    _Float16* encb  = (_Float16*)carve((size_t)NSEQ * LSEQ * DIN * 2);
    float*    curb  = (float*)carve((size_t)NSEQ * LSEQ * DM * 4);
    _Float16* qbuf  = (_Float16*)carve((size_t)NSEQ * TWO_L * DM * 2);
    _Float16* kbuf  = (_Float16*)carve((size_t)NSEQ * TWO_L * DM * 2);
    _Float16* vbuf  = (_Float16*)carve((size_t)NSEQ * TWO_L * DM * 2);
    float*    featb = (float*)carve((size_t)NSEQ * DFEAT * 4);

    (void)hipFuncSetAttribute((const void*)attnhp_attn,
                              hipFuncAttributeMaxDynamicSharedMemorySize, ATT_SMEM);
    (void)hipFuncSetAttribute((const void*)attnhp_qkv,
                              hipFuncAttributeMaxDynamicSharedMemorySize, QKV_SMEM);

    // weight conversion (transposed f16)
    attnhp_cvt_tr<<<(EMB3 * DM + 255) / 256, 256, 0, stream>>>(in_w, wInWt, EMB3, DM);
    for (int s = 0; s < 4; ++s) {
        attnhp_cvt_tr<<<(DIN * DM + 255) / 256, 256, 0, stream>>>(
            q_w + (size_t)s * DIN * DM, wQt + (size_t)s * DM * DIN, DIN, DM);
        attnhp_cvt_tr<<<(DIN * DM + 255) / 256, 256, 0, stream>>>(
            k_w + (size_t)s * DIN * DM, wKt + (size_t)s * DM * DIN, DIN, DM);
        attnhp_cvt_tr<<<(DIN * DM + 255) / 256, 256, 0, stream>>>(
            v_w + (size_t)s * DIN * DM, wVt + (size_t)s * DM * DIN, DIN, DM);
    }
    attnhp_cvt_tr<<<(DFEAT * DM + 255) / 256, 256, 0, stream>>>(w1, wW1t, DFEAT, DM);

    // embedding + input projection + temporal embedding
    attnhp_embed<<<NSEQ * LSEQ / 16, 128, 0, stream>>>(subj, obj, rel, timev, ent, relemb,
                                                       wInWt, in_b, encb, temb);

    // heads are serially chained through enc_input in the reference
    for (int h = 0; h < 2; ++h) {
        hipMemsetAsync(curb, 0, (size_t)NSEQ * LSEQ * DM * 4, stream);
        for (int l = 0; l < 2; ++l) {
            int s = h * 2 + l;
            attnhp_qkv<<<NSEQ * 16, 256, QKV_SMEM, stream>>>(
                encb, curb, temb,
                wQt + (size_t)s * DM * DIN, wKt + (size_t)s * DM * DIN,
                wVt + (size_t)s * DM * DIN,
                q_b + (size_t)s * DM, k_b + (size_t)s * DM, v_b + (size_t)s * DM,
                qbuf, kbuf, vbuf);
            attnhp_attn<<<NSEQ, 256, ATT_SMEM, stream>>>(qbuf, kbuf, vbuf, timev, encb, curb);
        }
        attnhp_head_out<<<(NSEQ * DM + 255) / 256, 256, 0, stream>>>(curb, featb, h);
    }

    attnhp_init_out<<<2, 256, 0, stream>>>(out, b2, NSEQ);
    attnhp_mlp<<<NSEQ / 16, 128, 0, stream>>>(featb, wW1t, b1, w2, out);
}
